// OPTNET_72705206387266
// MI455X (gfx1250) — matrix operand order repaired and checked by reading, not compiled, not run on hardware
//
#include <hip/hip_runtime.h>
#include <hip/hip_bf16.h>

// ---------------------------------------------------------------------------
// Batched QP projection via ADMM, restructured for MI455X (gfx1250):
//  * A stored as f16 in two layouts (row-major + transposed) -> 256 MB total,
//    streamed per iteration; K^-1 held in LDS; iteration loop is a single
//    persistent kernel (one workgroup per batch, no global sync needed).
//  * AAT = A A^T computed with v_wmma_f32_16x16x32_f16.
//  * K^-1 via in-LDS Gauss-Jordan (SPD, cond ~9, no pivoting needed).
// ---------------------------------------------------------------------------

typedef __attribute__((ext_vector_type(16))) _Float16 v16h;
typedef __attribute__((ext_vector_type(8)))  _Float16 v8h;
typedef __attribute__((ext_vector_type(4)))  _Float16 v4h;
typedef __attribute__((ext_vector_type(8)))  float    v8f;

#define BB   256
#define MM   256
#define VV   1024
#define KPAD 257         // LDS row stride for 256-wide rows (bank-conflict free)
#define JIT  1e-6f
#define NIT  100

__device__ __forceinline__ float wave_red(float a) {
    #pragma unroll
    for (int off = 16; off > 0; off >>= 1) a += __shfl_xor(a, off, 32);
    return a;
}

// ---------------------------------------------------------------------------
// Kernel 1: A (f32) -> Ah (f16 row-major [B,M,V]) and AhT (f16 [B,V,M])
// ---------------------------------------------------------------------------
__global__ void OPTNET_cvt_kernel(const float* __restrict__ A,
                                  _Float16* __restrict__ Ah,
                                  _Float16* __restrict__ AhT) {
    size_t idx  = (size_t)blockIdx.x * blockDim.x + threadIdx.x;
    size_t base = idx * 4;
    if (base >= (size_t)BB * MM * VV) return;
    const float4 f = *(const float4*)(A + base);
    _Float16 h0 = (_Float16)f.x, h1 = (_Float16)f.y;
    _Float16 h2 = (_Float16)f.z, h3 = (_Float16)f.w;
    v4h hv; hv[0] = h0; hv[1] = h1; hv[2] = h2; hv[3] = h3;
    *(v4h*)(Ah + base) = hv;                       // coalesced 8B store
    // transpose scatter (one-time cost)
    size_t v  = base % VV;
    size_t bm = base / VV;
    size_t m  = bm % MM;
    size_t b  = bm / MM;
    _Float16* tp = AhT + ((b * VV + v) * MM + m);
    tp[0]      = h0;
    tp[MM]     = h1;
    tp[2 * MM] = h2;
    tp[3 * MM] = h3;
}

// ---------------------------------------------------------------------------
// Kernel 2: AAT[b] = Ah[b] * Ah[b]^T + JIT*I   via v_wmma_f32_16x16x32_f16
// grid = B*256 blocks of 1 wave (32 threads); block computes one 16x16 tile.
// A fragment (16x32 f16, lane L<16: M=L, K = k0..k0+15;
//             lane L+16: same M, K = k0+16..31)
// B fragment (32x16 f16, lane n<16: N=n, K = k0..k0+15; lane n+16: K=k0+16..31)
// For AAT, B[k][n] = Ah[jrow=j0+n][k] -> contiguous 32B per lane.
// ---------------------------------------------------------------------------
__global__ void OPTNET_aat_kernel(const _Float16* __restrict__ Ah,
                                  float* __restrict__ AAT) {
    const int lane = threadIdx.x;          // 0..31
    const int bid  = blockIdx.x;
    const int b    = bid >> 8;
    const int ti   = (bid >> 4) & 15;
    const int tj   = bid & 15;
    const int rsel = lane >> 4;            // 0 or 1
    const int rlo  = lane & 15;

    const _Float16* arow = Ah + ((size_t)(b * MM + ti * 16 + rlo)) * VV;
    const _Float16* brow = Ah + ((size_t)(b * MM + tj * 16 + rlo)) * VV;

    v8f acc = {};
    for (int k0 = 0; k0 < VV; k0 += 32) {
        const _Float16* pa = arow + k0 + rsel * 16;
        v8h alo = *(const v8h*)(pa);
        v8h ahi = *(const v8h*)(pa + 8);
        v16h afrag;
        #pragma unroll
        for (int i = 0; i < 8; ++i) { afrag[i] = alo[i]; afrag[i + 8] = ahi[i]; }
        v16h bfrag = *(const v16h*)(brow + k0 + rsel * 16);
        acc = __builtin_amdgcn_wmma_f32_16x16x32_f16(
                  false, afrag, false, bfrag, (short)0, acc, false, false);
    }
    // C/D layout: VGPR g -> M = g + 8*rsel, N = rlo
    const int gn = tj * 16 + rlo;
    #pragma unroll
    for (int g = 0; g < 8; ++g) {
        const int gm = ti * 16 + g + rsel * 8;
        float vv = acc[g];
        if (gm == gn) vv += JIT;
        AAT[((size_t)b * MM + gm) * MM + gn] = vv;
    }
}

// ---------------------------------------------------------------------------
// Kernel 3: Kinv[b] = inverse(AAT[b]) — Gauss-Jordan in LDS, padded rows.
// 256 threads per block; thread tid owns row tid.
// ---------------------------------------------------------------------------
__global__ void OPTNET_gj_kernel(const float* __restrict__ AAT,
                                 float* __restrict__ Kinv) {
    extern __shared__ float lds[];         // 256 * 257 floats
    const int tid = threadIdx.x;
    const int b   = blockIdx.x;
    const float* src = AAT + (size_t)b * MM * MM;
    for (int i = 0; i < MM; ++i) lds[i * KPAD + tid] = src[i * MM + tid];
    __syncthreads();

    for (int p = 0; p < MM; ++p) {
        const float pinv = 1.0f / lds[p * KPAD + p];   // broadcast read
        const float g    = lds[tid * KPAD + p] * pinv; // own col-p element
        __syncthreads();
        if (tid != p) {
            float*       row  = lds + tid * KPAD;
            const float* prow = lds + p * KPAD;
            #pragma unroll 4
            for (int j = 0; j < MM; ++j) row[j] -= g * prow[j];
            row[p] = -g;
        }
        __syncthreads();
        // scale pivot row in parallel (thread tid handles column tid)
        lds[p * KPAD + tid] = (tid == p) ? pinv : lds[p * KPAD + tid] * pinv;
        __syncthreads();
    }
    float* dst = Kinv + (size_t)b * MM * MM;
    for (int i = 0; i < MM; ++i) dst[i * MM + tid] = lds[i * KPAD + tid];
}

// ---------------------------------------------------------------------------
// Kernel 4: persistent ADMM. One block (256 thr = 8 waves) per batch.
// LDS: Kinv (padded, 257KB) + r/y (4KB) + t (1KB) + s (1KB).
// Per iteration: t = Ah r ; s = Kinv t (LDS) ; y = AhT s ;
//                xk = 0.5*(r - y) + d ; ADMM elementwise update.
// ---------------------------------------------------------------------------
__global__ void OPTNET_admm_kernel(const float* __restrict__ x0,
                                   const float* __restrict__ bvec,
                                   const int*   __restrict__ mask,
                                   const _Float16* __restrict__ Ah,
                                   const _Float16* __restrict__ AhT,
                                   const float* __restrict__ Kinv,
                                   float* __restrict__ out) {
    extern __shared__ float smem[];
    float* kv   = smem;                    // MM*KPAD
    float* rbuf = kv + MM * KPAD;          // VV  (holds r, then y)
    float* tb   = rbuf + VV;               // MM
    float* sb   = tb + MM;                 // MM

    const int tid = threadIdx.x;
    const int b   = blockIdx.x;
    const int wv  = tid >> 5;
    const int ln  = tid & 31;
    const int v0  = tid * 4;

    // resident K^-1
    const float* kp = Kinv + (size_t)b * MM * MM;
    for (int i = 0; i < MM; ++i) kv[i * KPAD + tid] = kp[i * MM + tid];

    // per-thread state (4 variables each)
    float xr[4], zr[4], ur[4], dr[4];
    int   mk[4];
    #pragma unroll
    for (int j = 0; j < 4; ++j) {
        xr[j] = x0[(size_t)b * VV + v0 + j];
        zr[j] = 0.0f; ur[j] = 0.0f;
        mk[j] = mask[v0 + j];
    }

    const _Float16* abase = Ah  + (size_t)b * MM * VV;
    const _Float16* tbase = AhT + (size_t)b * VV * MM;

    // ---- preamble: kb = Kinv*b ; d = A^T kb ----
    tb[tid] = bvec[(size_t)b * MM + tid];
    __syncthreads();
    for (int mi = 0; mi < 32; ++mi) {
        const int m = wv * 32 + mi;
        const float* krow = kv + m * KPAD;
        float acc = 0.0f;
        #pragma unroll
        for (int k = 0; k < 8; ++k) acc += krow[ln + k * 32] * tb[ln + k * 32];
        acc = wave_red(acc);
        if (ln == 0) sb[m] = acc;
    }
    __syncthreads();
    for (int vi = 0; vi < 128; ++vi) {
        const int v = wv * 128 + vi;
        v8h av = *(const v8h*)(tbase + (size_t)v * MM + ln * 8);
        float acc = 0.0f;
        #pragma unroll
        for (int i = 0; i < 8; ++i) acc += (float)av[i] * sb[ln * 8 + i];
        acc = wave_red(acc);
        if (ln == 0) rbuf[v] = acc;
    }
    __syncthreads();
    #pragma unroll
    for (int j = 0; j < 4; ++j) dr[j] = rbuf[v0 + j];
    #pragma unroll
    for (int j = 0; j < 4; ++j) rbuf[v0 + j] = xr[j];   // r0 = x (z=u=0)

    // ---- 100 ADMM iterations + final update ----
    for (int it = 0; it <= NIT; ++it) {
        __syncthreads();                    // r ready in rbuf
        // t = Ah * r
        for (int mi = 0; mi < 32; ++mi) {
            const int m = wv * 32 + mi;
            const _Float16* arow = abase + (size_t)m * VV;
            float acc = 0.0f;
            #pragma unroll
            for (int c = 0; c < 4; ++c) {
                v8h av = *(const v8h*)(arow + c * 256 + ln * 8);
                #pragma unroll
                for (int i = 0; i < 8; ++i)
                    acc += (float)av[i] * rbuf[c * 256 + ln * 8 + i];
            }
            acc = wave_red(acc);
            if (ln == 0) tb[m] = acc;
        }
        __syncthreads();
        // s = Kinv * t   (all from LDS)
        for (int mi = 0; mi < 32; ++mi) {
            const int m = wv * 32 + mi;
            const float* krow = kv + m * KPAD;
            float acc = 0.0f;
            #pragma unroll
            for (int k = 0; k < 8; ++k) acc += krow[ln + k * 32] * tb[ln + k * 32];
            acc = wave_red(acc);
            if (ln == 0) sb[m] = acc;
        }
        __syncthreads();
        // y = AhT * s   (overwrite rbuf with y; r is recomputed from regs)
        for (int vi = 0; vi < 128; ++vi) {
            const int v = wv * 128 + vi;
            v8h av = *(const v8h*)(tbase + (size_t)v * MM + ln * 8);
            float acc = 0.0f;
            #pragma unroll
            for (int i = 0; i < 8; ++i) acc += (float)av[i] * sb[ln * 8 + i];
            acc = wave_red(acc);
            if (ln == 0) rbuf[v] = acc;
        }
        __syncthreads();
        // elementwise: xk = 0.5*(r - y) + d ; project ; update z,u ; next r
        if (it == NIT) {
            #pragma unroll
            for (int j = 0; j < 4; ++j) {
                const float r  = xr[j] + zr[j] - ur[j];
                const float xk = 0.5f * (r - rbuf[v0 + j]) + dr[j];
                out[(size_t)b * VV + v0 + j] = xk;
            }
        } else {
            #pragma unroll
            for (int j = 0; j < 4; ++j) {
                const float r  = xr[j] + zr[j] - ur[j];
                const float xk = 0.5f * (r - rbuf[v0 + j]) + dr[j];
                const float w  = xk + ur[j];
                const float z  = (mk[j] > 0) ? fminf(w, 0.0f) : w;
                ur[j] = w - z;
                zr[j] = z;
                rbuf[v0 + j] = xr[j] + z - ur[j];
            }
        }
    }
}

// ---------------------------------------------------------------------------
extern "C" void kernel_launch(void* const* d_in, const int* in_sizes, int n_in,
                              void* d_out, int out_size, void* d_ws, size_t ws_size,
                              hipStream_t stream) {
    const float* x    = (const float*)d_in[0];   // [B,V]
    const float* bvec = (const float*)d_in[1];   // [B,M]
    const float* A    = (const float*)d_in[2];   // [B,M,V]
    const int*   mask = (const int*)d_in[3];     // [V]
    float* out = (float*)d_out;                  // [B,V]

    const size_t nA = (size_t)BB * MM * VV;      // 67,108,864 elements
    char* ws = (char*)d_ws;
    _Float16* Ah   = (_Float16*)(ws);                         // 128 MB
    _Float16* AhT  = (_Float16*)(ws + nA * 2);                // 128 MB
    float*    AAT  = (float*)(ws + nA * 4);                   // 64 MB
    float*    Kinv = (float*)(ws + nA * 4 + (size_t)BB * MM * MM * 4); // 64 MB

    // 1) convert / transpose
    {
        const size_t threads = nA / 4;           // 4 elems per thread
        OPTNET_cvt_kernel<<<(unsigned)(threads / 256), 256, 0, stream>>>(A, Ah, AhT);
    }
    // 2) WMMA SYRK: AAT = Ah Ah^T + eps I
    OPTNET_aat_kernel<<<BB * 256, 32, 0, stream>>>(Ah, AAT);
    // 3) batched inverse in LDS
    OPTNET_gj_kernel<<<BB, 256, MM * KPAD * sizeof(float), stream>>>(AAT, Kinv);
    // 4) persistent ADMM (K^-1 resident in LDS)
    {
        const size_t shmem = (size_t)(MM * KPAD + VV + MM + MM) * sizeof(float);
        OPTNET_admm_kernel<<<BB, 256, shmem, stream>>>(x, bvec, mask, Ah, AhT, Kinv, out);
    }
}
